// TransitionLayer_33921651704540
// MI455X (gfx1250) — matrix operand order repaired and checked by reading, not compiled, not run on hardware
//
#include <hip/hip_runtime.h>

// MI455X / gfx1250: wave32, v_wmma_f32_16x16x32_bf16, fp32 accumulate.
// BN+ReLU+1x1conv+2x2avgpool fused. GEMM O(256) x K(512) x M(200704),
// bf16 hi/lo split (3 WMMAs) ~ fp32 accuracy on the fast bf16 matrix pipe.
// One 512-thread WG per (n, pooled row): x staged once (min HBM traffic),
// double-buffered LDS staging, 1 barrier/chunk, fully affine addressing.

typedef __attribute__((ext_vector_type(16))) __bf16 v16bf;
typedef __attribute__((ext_vector_type(8)))  float  v8f;

union Frag {               // 32 bytes: one WMMA A/B fragment (16 bf16)
    v16bf v;
    uint4 q[2];
};

#define EPSF 1e-5f
// Per pixel-column: 32 bf16 K-slots padded to 40 (80 B = 20-dword stride:
// 16B-aligned fragment starts, conflict-free bank spread for b128 reads).
#define PIX_STRIDE 40

__global__ __launch_bounds__(512) void transition_kernel(
    const float* __restrict__ x,      // [64,512,56,56]
    const float* __restrict__ bn_w,   // [512]
    const float* __restrict__ bn_b,   // [512]
    const float* __restrict__ bn_m,   // [512]
    const float* __restrict__ bn_v,   // [512]
    const float* __restrict__ cw,     // [256,512]
    float* __restrict__ out)          // [64,256,28,28]
{
    __shared__ float2 s_bn[512];      // (scale, shift) folded BN
    __shared__ union {
        __bf16 stage[2][2][112 * PIX_STRIDE]; // [buf][hi/lo][p*40 + kc]
        float  conv[256 * 112];               // epilogue conv tile
    } sm;

    const int tid  = threadIdx.x;
    const int lane = tid & 31;
    const int wid  = tid >> 5;     // 0..15 : wave's 16-wide O tile
    const int col  = lane & 15;    // WMMA row (A) / column (B,C,D)
    const int half = lane >> 4;    // lane half selects K sub-range

    const int h1 = blockIdx.x;     // pooled row 0..27
    const int n  = blockIdx.y;     // batch 0..63

    // ---- fold BN into per-channel (scale, shift): one channel/thread ----
    {
        int c = tid;
        float sc = bn_w[c] * rsqrtf(bn_v[c] + EPSF);
        s_bn[c] = make_float2(sc, bn_b[c] - bn_m[c] * sc);
    }
    __syncthreads();

    // Staging role: fixed channel-in-chunk kc per thread, pixels p0+16i.
    // Rows 2h1 and 2h1+1 are contiguous: x off = c*3136 + 112*h1 + p.
    const int kc = tid >> 4;              // 0..31
    const int p0 = tid & 15;              // pixel phase
    const int stBase = p0 * PIX_STRIDE + kc;  // LDS slot (element index)
    const float* __restrict__ xp =
        x + (size_t)n * (512 * 3136) + (size_t)h1 * 112 + (size_t)kc * 3136 + p0;

    // A-matrix row for this lane (weights are L2-resident: 512 KB total).
    // A k order used here: K = 16*half + e (contiguous, matches B staging).
    const int oRow = wid * 16 + col;
    const float* __restrict__ wrow = cw + (size_t)oRow * 512 + half * 16;

    v8f acc[7];
#pragma unroll
    for (int t = 0; t < 7; ++t) { v8f z = {}; acc[t] = z; }

    // ---- prologue: stage chunk 0 into buffer 0 ----
    {
        float xv[7];
#pragma unroll
        for (int i = 0; i < 7; ++i) xv[i] = xp[16 * i];
        float2 ss = s_bn[kc];
#pragma unroll
        for (int i = 0; i < 7; ++i) {
            float a  = fmaxf(xv[i] * ss.x + ss.y, 0.0f);
            __bf16 hi = (__bf16)a;
            __bf16 lo = (__bf16)(a - (float)hi);
            sm.stage[0][0][stBase + 640 * i] = hi;
            sm.stage[0][1][stBase + 640 * i] = lo;
        }
    }
    __syncthreads();

    // ---- main loop: compute buf cur, stage next chunk into buf cur^1 ----
    for (int ck = 0; ck < 16; ++ck) {
        const int cur = ck & 1;

        // A-fragment loads first (ready at loadcnt <= #x-loads)...
        const float* wp = wrow + ck * 32;
        float4 w0 = *(const float4*)(wp);
        float4 w1 = *(const float4*)(wp + 4);
        float4 w2 = *(const float4*)(wp + 8);
        float4 w3 = *(const float4*)(wp + 12);

        // ...then next chunk's HBM loads: latency hides under the WMMAs.
        float  xn[7];
        float2 ssn;
        if (ck < 15) {
            const float* xq = xp + (size_t)(ck + 1) * (32 * 3136);
#pragma unroll
            for (int i = 0; i < 7; ++i) xn[i] = xq[16 * i];
            ssn = s_bn[(ck + 1) * 32 + kc];
        }

        Frag ahi, alo;
        {
            float wf[16] = {w0.x, w0.y, w0.z, w0.w, w1.x, w1.y, w1.z, w1.w,
                            w2.x, w2.y, w2.z, w2.w, w3.x, w3.y, w3.z, w3.w};
#pragma unroll
            for (int e = 0; e < 16; ++e) {
                __bf16 h = (__bf16)wf[e];
                ahi.v[e] = h;
                alo.v[e] = (__bf16)(wf[e] - (float)h);
            }
        }

        // 7 N-tiles x (hi*hi + hi*lo + lo*hi); C-chained WMMAs are the
        // architected accumulate pattern (no D->C hazard in the ISA table).
        const char* sb = (const char*)&sm.stage[cur][0][0]
                         + (size_t)col * 80 + (size_t)half * 32;
#pragma unroll
        for (int t = 0; t < 7; ++t) {
            const char* bp = sb + t * (16 * 80);
            Frag bhi, blo;
            bhi.q[0] = *(const uint4*)(bp);
            bhi.q[1] = *(const uint4*)(bp + 16);
            blo.q[0] = *(const uint4*)(bp + 8960);   // lo plane
            blo.q[1] = *(const uint4*)(bp + 8976);
            acc[t] = __builtin_amdgcn_wmma_f32_16x16x32_bf16(
                false, ahi.v, false, bhi.v, (short)0, acc[t], false, false);
            acc[t] = __builtin_amdgcn_wmma_f32_16x16x32_bf16(
                false, ahi.v, false, blo.v, (short)0, acc[t], false, false);
            acc[t] = __builtin_amdgcn_wmma_f32_16x16x32_bf16(
                false, alo.v, false, bhi.v, (short)0, acc[t], false, false);
        }

        // BN+ReLU+split+store the prefetched chunk into the other buffer.
        if (ck < 15) {
            __bf16* d0 = &sm.stage[cur ^ 1][0][0];
            __bf16* d1 = &sm.stage[cur ^ 1][1][0];
#pragma unroll
            for (int i = 0; i < 7; ++i) {
                float a  = fmaxf(xn[i] * ssn.x + ssn.y, 0.0f);
                __bf16 hi = (__bf16)a;
                __bf16 lo = (__bf16)(a - (float)hi);
                d0[stBase + 640 * i] = hi;
                d1[stBase + 640 * i] = lo;
            }
        }
        __syncthreads();  // next buffer complete; cur reusable next+1 iter
    }

    // ---- epilogue: spill conv tile to LDS (union reuses staging space;
    //      the loop's final barrier guarantees staging reads are done) ----
#pragma unroll
    for (int t = 0; t < 7; ++t) {
#pragma unroll
        for (int r = 0; r < 8; ++r) {
            int o_local = wid * 16 + r + half * 8;   // C/D layout: M = r + 8*half
            sm.conv[o_local * 112 + t * 16 + col] = acc[t][r];
        }
    }
    __syncthreads();

    // ---- 2x2 avg pool + coalesced store: 256 O x 28 w' = 14 per thread ----
    const size_t obase = (size_t)n * (256 * 784) + (size_t)h1 * 28;
#pragma unroll
    for (int i = 0; i < 14; ++i) {
        int idx = tid + i * 512;          // 0..7167
        int o   = idx / 28;
        int wq  = idx - o * 28;
        const float* r0 = &sm.conv[o * 112 + 2 * wq];
        float v = 0.25f * (r0[0] + r0[1] + r0[56] + r0[57]);
        out[obase + (size_t)o * 784 + wq] = v;
    }
}

extern "C" void kernel_launch(void* const* d_in, const int* in_sizes, int n_in,
                              void* d_out, int out_size, void* d_ws, size_t ws_size,
                              hipStream_t stream) {
    (void)in_sizes; (void)n_in; (void)out_size; (void)d_ws; (void)ws_size;
    const float* x    = (const float*)d_in[0];
    const float* bn_w = (const float*)d_in[1];
    const float* bn_b = (const float*)d_in[2];
    const float* bn_m = (const float*)d_in[3];
    const float* bn_v = (const float*)d_in[4];
    const float* cw   = (const float*)d_in[5];
    float* out = (float*)d_out;

    dim3 grid(28, 64);   // (pooled row h', batch n)
    transition_kernel<<<grid, 512, 0, stream>>>(x, bn_w, bn_b, bn_m, bn_v, cw, out);
}
